// PMAT_3676492005785
// MI455X (gfx1250) — matrix operand order repaired and checked by reading, not compiled, not run on hardware
//
#include <hip/hip_runtime.h>
#include <hip/hip_bf16.h>
#include <math.h>

typedef __attribute__((ext_vector_type(16))) _Float16 v16h;
typedef __attribute__((ext_vector_type(8)))  float    v8f;

#define DIM 128

// ---------------------------------------------------------------- zero scratch
__global__ void PMAT_zero_kernel(float* __restrict__ p, int n) {
    int i = blockIdx.x * blockDim.x + threadIdx.x;
    if (i < n) p[i] = 0.0f;
}

// ------------------------------------------------- l2-normalize (+ optional noise)
// one wave (32 lanes) per row; 4 floats per lane (128 = 32*4)
__global__ void PMAT_normalize_kernel(const float* __restrict__ in,
                                      const float* __restrict__ noise, // may be null
                                      float* __restrict__ out, int N) {
    int row  = blockIdx.x * (blockDim.x >> 5) + (threadIdx.x >> 5);
    int lane = threadIdx.x & 31;
    if (row >= N) return;
    size_t base = (size_t)row * DIM + lane * 4;
    float4 v = *(const float4*)(in + base);
    if (noise) {
        float4 nz = *(const float4*)(noise + base);
        v.x += nz.x; v.y += nz.y; v.z += nz.z; v.w += nz.w;
    }
    float ss = v.x*v.x + v.y*v.y + v.z*v.z + v.w*v.w;
    #pragma unroll
    for (int off = 16; off > 0; off >>= 1)
        ss += __shfl_xor(ss, off, 32);
    float nrm = sqrtf(ss);
    float s = 1.0f / fmaxf(nrm, 1e-12f);
    float4 o = { v.x * s, v.y * s, v.z * s, v.w * s };
    *(float4*)(out + base) = o;
}

// ------------------------------------------------------- per-node projections (WMMA)
// p[i] = dot(h[i], W[k, 0:128]);  q[i] = dot(h[i], W[k, 128:256])
// One wave per 16-row tile. D = A(16x32 f16) * B(32x16 f16) + C, K accumulated 128/32.
// B column 0 = W-first-half, column 1 = W-second-half, columns 2..15 = 0.
__global__ void PMAT_proj_kernel(const float* __restrict__ h,   // [N,128]
                                 const float* __restrict__ Wk,  // [256]
                                 float* __restrict__ p,
                                 float* __restrict__ q, int N) {
    int tile = blockIdx.x;
    int lane = threadIdx.x;            // 0..31 (block is exactly one wave)
    int mrow = tile * 16 + (lane & 15);
    int rcl  = mrow < N ? mrow : N - 1;          // clamp reads; writes guarded below
    int khalf = (lane >> 4) * 8;                 // lanes 0-15 -> 0, lanes 16-31 -> 8
    int col   = lane & 15;

    v8f c = {};
    #pragma unroll
    for (int kb = 0; kb < DIM; kb += 32) {
        // ---- A tile: 16x32 f16. Lane layout per ISA:
        //   lanes 0-15 : row M=lane,   K = kb+{0..7, 16..23}
        //   lanes 16-31: row M=lane-16,K = kb+{8..15, 24..31}
        const float* arow = h + (size_t)rcl * DIM + kb + khalf;
        v16h a;
        #pragma unroll
        for (int t = 0; t < 8; ++t) a[t]     = (_Float16)arow[t];
        #pragma unroll
        for (int t = 0; t < 8; ++t) a[8 + t] = (_Float16)arow[16 + t];

        // ---- B tile: 32x16 f16. Lane n<16: col n, K=kb+{0..7,16..23}; lane n+16: K=kb+{8..15,24..31}
        int kb2 = kb + ((lane >> 4) * 8);
        v16h bm;
        if (col < 2) {
            const float* wc = Wk + col * DIM + kb2;
            #pragma unroll
            for (int t = 0; t < 8; ++t) bm[t] = (_Float16)wc[t];
            #pragma unroll
            for (int t = 0; t < 8; ++t) bm[8 + t] = (_Float16)wc[16 + t];
        } else {
            #pragma unroll
            for (int t = 0; t < 16; ++t) bm[t] = (_Float16)0.0f;
        }

        c = __builtin_amdgcn_wmma_f32_16x16x32_f16(
                /*neg_a=*/false, a, /*neg_b=*/false, bm,
                /*c_mod=*/(short)0, c, /*reuse_a=*/false, /*reuse_b=*/false);
    }

    // C layout: c[i] = (M = 8*(lane>>4) + i, N = lane&15)
    int mbase = tile * 16 + ((lane >> 4) * 8);
    if (col == 0) {
        #pragma unroll
        for (int i = 0; i < 8; ++i) { int r = mbase + i; if (r < N) p[r] = c[i]; }
    } else if (col == 1) {
        #pragma unroll
        for (int i = 0; i < 8; ++i) { int r = mbase + i; if (r < N) q[r] = c[i]; }
    }
}

// ------------------------------------------------------------------ edge scatter
// One wave per edge (32 lanes x float4 = 128 floats). alpha = sigmoid(p[s]+q[d]+b[k]).
// aggr[dst] += alpha * h[src]  via L2-resident global_atomic_add_f32.
__global__ void PMAT_edge_kernel(const int* __restrict__ src,
                                 const int* __restrict__ dst,
                                 const float* __restrict__ h,
                                 const float* __restrict__ p,
                                 const float* __restrict__ q,
                                 const float* __restrict__ b, int k,
                                 float* __restrict__ aggr, long long E) {
    long long t = (long long)blockIdx.x * blockDim.x + threadIdx.x;
    long long e = t >> 5;
    if (e >= E) return;
    int lane = (int)(t & 31);
    int s = src[e];
    int d = dst[e];
    float logit = p[s] + q[d] + b[k];
    float alpha = 1.0f / (1.0f + __expf(-logit));
    const float4 hv = *(const float4*)(h + (size_t)s * DIM + lane * 4);
    float* ad = aggr + (size_t)d * DIM + lane * 4;
    unsafeAtomicAdd(ad + 0, alpha * hv.x);
    unsafeAtomicAdd(ad + 1, alpha * hv.y);
    unsafeAtomicAdd(ad + 2, alpha * hv.z);
    unsafeAtomicAdd(ad + 3, alpha * hv.w);
}

extern "C" void kernel_launch(void* const* d_in, const int* in_sizes, int n_in,
                              void* d_out, int out_size, void* d_ws, size_t ws_size,
                              hipStream_t stream) {
    const float* x     = (const float*)d_in[0];   // [N,128]
    const int*   eidx  = (const int*)  d_in[1];   // [2,E]
    const float* W     = (const float*)d_in[2];   // [H,256]
    const float* b     = (const float*)d_in[3];   // [H]
    const float* noise = (const float*)d_in[4];   // [H,N,128]

    const int       N = in_sizes[0] / DIM;        // 40000
    const long long E = in_sizes[1] / 2;          // 640000
    const int       H = in_sizes[3];              // 3

    float* out  = (float*)d_out;                  // [(H+1), N, 128]
    float* aggr = (float*)d_ws;                   // [N,128]
    float* p    = aggr + (size_t)N * DIM;         // [N]
    float* q    = p + N;                          // [N]

    const int*  src = eidx;
    const int*  dst = eidx + E;
    const size_t slab = (size_t)N * DIM;

    const int normBlocks = (N + 7) / 8;           // 8 rows (waves) per 256-thr block
    const int zeroBlocks = (int)((slab + 255) / 256);
    const int projTiles  = (N + 15) / 16;
    const long long edgeThreads = E * 32;
    const int edgeBlocks = (int)((edgeThreads + 255) / 256);

    // out[0] = l2norm(x)
    PMAT_normalize_kernel<<<normBlocks, 256, 0, stream>>>(x, nullptr, out, N);

    for (int k = 0; k < H; ++k) {
        const float* hcur = out + (size_t)k * slab;
        float*       hnew = out + (size_t)(k + 1) * slab;

        PMAT_zero_kernel<<<zeroBlocks, 256, 0, stream>>>(aggr, (int)slab);

        PMAT_proj_kernel<<<projTiles, 32, 0, stream>>>(hcur, W + (size_t)k * 2 * DIM,
                                                       p, q, N);

        PMAT_edge_kernel<<<edgeBlocks, 256, 0, stream>>>(src, dst, hcur, p, q,
                                                         b, k, aggr, E);

        PMAT_normalize_kernel<<<normBlocks, 256, 0, stream>>>(aggr, noise + (size_t)k * slab,
                                                              hnew, N);
    }
}
